// trajectory2seq_26989574488434
// MI455X (gfx1250) — matrix operand order repaired and checked by reading, not compiled, not run on hardware
//
#include <hip/hip_runtime.h>

// ---------------- problem constants ----------------
#define HID  512
#define BSZ  64
#define TSEQ 512
#define VOC  30
#define MAXL 64
#define G3H  1536                   // 3*HID
#define WSTR (G3H*HID)              // elements per weight matrix (1536x512)
#define BMV  (BSZ*MAXL*VOC)         // 122880 logits

// ---------------- persistent-kernel config ----------------
#define NBLK 24
#define TPB  256
#define NWAVES ((NBLK*TPB)/32)      // 192 wave32 waves

typedef __bf16 bf16;
typedef bf16  v16bf __attribute__((ext_vector_type(16)));
typedef bf16  v8bf  __attribute__((ext_vector_type(8)));
typedef float v8f   __attribute__((ext_vector_type(8)));

// ---------------- small helpers ----------------
__device__ __forceinline__ float bf2f(unsigned short b) {
    return __uint_as_float(((unsigned)b) << 16);
}
__device__ __forceinline__ unsigned short f2bf(float f) {
    unsigned u = __float_as_uint(f);
    unsigned r = (u + 0x7fffu + ((u >> 16) & 1u)) >> 16;   // round-to-nearest-even
    return (unsigned short)r;
}

// Device-wide barrier: every thread fences its stores, block arrives once,
// thread0 spins (with s_sleep) on a monotonically increasing counter.
__device__ __forceinline__ void grid_sync(unsigned* cnt, unsigned& phase) {
    __threadfence();
    __syncthreads();
    if (threadIdx.x == 0) {
        const unsigned target = (phase + 1u) * (unsigned)NBLK;
        atomicAdd(cnt, 1u);
        while (((volatile unsigned*)cnt)[0] < target)
            __builtin_amdgcn_s_sleep(2);
    }
    __syncthreads();
    __threadfence();   // acquire: invalidate stale near-cache lines
    phase++;
}

// Cooperatively copy the 64KB activation matrix A[64,512] (bf16) into LDS.
// Leading barrier guards WAR against the previous GEMM's LDS reads.
__device__ __forceinline__ void stage_A_lds(const unsigned short* __restrict__ Ab,
                                            unsigned short* lds) {
    __syncthreads();
    for (int i = threadIdx.x * 8; i < BSZ * HID; i += TPB * 8)
        *(v8bf*)(lds + i) = *(const v8bf*)(Ab + i);
    __syncthreads();
}

// One wave computes a 32x16 strip of  C[64,1536] = A[64,512] * W[1536,512]^T.
// A is read from LDS (staged by stage_A_lds); W streams from global/L2 and the
// compiler hoists all 16 K-step B-fragments into registers.
// task: bit0 = M-half (0 -> rows 0..31, 1 -> rows 32..63), task>>1 = N-strip.
typedef union { v16bf v; v8bf h[2]; } afrag_t;

__device__ __forceinline__ void gemm_strip2(
        const unsigned short* __restrict__ AldsU,  // LDS: [64,512] bf16 bits
        const unsigned short* __restrict__ Wb,     // [1536,512] bf16 bits
        float* __restrict__ Cg,                    // [64,1536] f32
        int task, int lane)
{
    const int nstrip = task >> 1;
    const int m0     = (task & 1) * 32;
    const int l15    = lane & 15;
    const int lh     = lane >> 4;
    const bf16* W = (const bf16*)Wb;
    const bf16* A = (const bf16*)AldsU;
    const bf16* wrow  = W + (nstrip * 16 + l15) * HID + lh * 16;
    const bf16* arow0 = A + (m0 + l15) * HID + lh * 16;
    const bf16* arow1 = arow0 + 16 * HID;

    v8f acc0 = {0.f,0.f,0.f,0.f,0.f,0.f,0.f,0.f};
    v8f acc1 = {0.f,0.f,0.f,0.f,0.f,0.f,0.f,0.f};

#pragma unroll
    for (int kk = 0; kk < 16; ++kk) {
        const int k0 = kk * 32;
        afrag_t a0, a1;
        a0.h[0] = *(const v8bf*)(arow0 + k0);
        a0.h[1] = *(const v8bf*)(arow0 + k0 + 8);
        a1.h[0] = *(const v8bf*)(arow1 + k0);
        a1.h[1] = *(const v8bf*)(arow1 + k0 + 8);
        v16bf bfrag = *(const v16bf*)(wrow + k0);
        acc0 = __builtin_amdgcn_wmma_f32_16x16x32_bf16(false, a0.v, false, bfrag,
                                                       (short)0, acc0, false, false);
        acc1 = __builtin_amdgcn_wmma_f32_16x16x32_bf16(false, a1.v, false, bfrag,
                                                       (short)0, acc1, false, false);
    }
    const int n = nstrip * 16 + l15;
#pragma unroll
    for (int r = 0; r < 8; ++r) {
        Cg[(m0 + lh * 8 + r)      * G3H + n] = acc0[r];
        Cg[(m0 + 16 + lh * 8 + r) * G3H + n] = acc1[r];
    }
}

// GRU gate combine: h = (1-z)*n + z*h ; biases folded in here (raw GEMM gates).
__device__ __forceinline__ void gru_combine(
        int tid, int nth,
        const float* __restrict__ gi, const float* __restrict__ gh,
        const float* __restrict__ bi, const float* __restrict__ bh,
        float* __restrict__ hf, unsigned short* __restrict__ hb,
        unsigned short* __restrict__ enc_outs /*nullable*/, int t)
{
    for (int e = tid; e < BSZ * HID; e += nth) {
        const int b = e >> 9, j = e & 511;
        const float* gib = gi + b * G3H;
        const float* ghb = gh + b * G3H;
        float gir = gib[j]          + bi[j];
        float giz = gib[512 + j]    + bi[512 + j];
        float gin = gib[1024 + j]   + bi[1024 + j];
        float ghr = ghb[j]          + bh[j];
        float ghz = ghb[512 + j]    + bh[512 + j];
        float ghn = ghb[1024 + j]   + bh[1024 + j];
        float r  = 1.f / (1.f + expf(-(gir + ghr)));
        float z  = 1.f / (1.f + expf(-(giz + ghz)));
        float nn = tanhf(gin + r * ghn);
        float hv = (1.f - z) * nn + z * hf[e];
        hf[e] = hv;
        unsigned short hq = f2bf(hv);
        hb[e] = hq;
        if (enc_outs) enc_outs[(b * TSEQ + t) * HID + j] = hq;
    }
}

// ---------------- init: bf16 weight conversion + state zeroing ----------------
__global__ void __launch_bounds__(TPB) t2s_init_kernel(
        const float* w0, const float* w1, const float* w2, const float* w3,
        const float* w4, const float* w5, const float* w6, const float* emb,
        unsigned short* Wb, unsigned short* embb,
        float* h0f, float* h1f, unsigned short* h0b, unsigned short* h1b,
        float* gA, int* tok)
{
    const int tid = blockIdx.x * TPB + threadIdx.x;
    const int nth = gridDim.x * TPB;
    const float* srcs[7] = {w0, w1, w2, w3, w4, w5, w6};
#pragma unroll 1
    for (int m = 0; m < 7; ++m)
        for (int i = tid; i < WSTR; i += nth)
            Wb[m * WSTR + i] = f2bf(srcs[m][i]);
    for (int i = tid; i < VOC * HID; i += nth) embb[i] = f2bf(emb[i]);
    for (int i = tid; i < BSZ * HID; i += nth) {
        h0f[i] = 0.f; h1f[i] = 0.f; h0b[i] = 0; h1b[i] = 0;
    }
    for (int i = tid; i < BSZ * G3H; i += nth) gA[i] = 0.f;   // gh0 for t=0
    for (int i = tid; i < BSZ; i += nth) tok[i] = 0;
}

// ---------------- encoder: persistent, 3 grid syncs per timestep ----------------
__global__ void __launch_bounds__(TPB) t2s_encoder_kernel(
        const float* __restrict__ x,
        const float* __restrict__ Wih0, const float* __restrict__ bih0,
        const float* __restrict__ bhh0,
        const unsigned short* __restrict__ Whh0b,
        const unsigned short* __restrict__ Wih1b,
        const unsigned short* __restrict__ Whh1b,
        const float* __restrict__ bih1, const float* __restrict__ bhh1,
        float* __restrict__ h0f, float* __restrict__ h1f,
        unsigned short* __restrict__ h0b, unsigned short* __restrict__ h1b,
        float* __restrict__ gA /*gh0*/, float* __restrict__ gB /*gh1*/,
        float* __restrict__ gC /*gi1*/,
        unsigned short* __restrict__ enc_outs,
        unsigned* __restrict__ cnt)
{
    __shared__ __align__(16) unsigned short ldsA[BSZ * HID];   // 64 KB
    const int tid  = blockIdx.x * TPB + threadIdx.x;
    const int nth  = NBLK * TPB;
    const int wave = tid >> 5, lane = tid & 31;
    unsigned phase = 0;

    for (int t = 0; t < TSEQ; ++t) {
        // Phase A: gh1(t) = h1 @ Whh1^T (A=h1 via LDS)  ||  combine layer0 -> h0(t)
        stage_A_lds(h1b, ldsA);
        gemm_strip2(ldsA, Whh1b, gB, wave, lane);
        for (int e = tid; e < BSZ * HID; e += nth) {
            const int b = e >> 9, j = e & 511;
            const float x0 = x[(b * TSEQ + t) * 2 + 0];
            const float x1 = x[(b * TSEQ + t) * 2 + 1];
            float gir = x0 * Wih0[j * 2]          + x1 * Wih0[j * 2 + 1]          + bih0[j];
            float giz = x0 * Wih0[(512 + j) * 2]  + x1 * Wih0[(512 + j) * 2 + 1]  + bih0[512 + j];
            float gin = x0 * Wih0[(1024 + j) * 2] + x1 * Wih0[(1024 + j) * 2 + 1] + bih0[1024 + j];
            float ghr = gA[b * G3H + j]        + bhh0[j];
            float ghz = gA[b * G3H + 512 + j]  + bhh0[512 + j];
            float ghn = gA[b * G3H + 1024 + j] + bhh0[1024 + j];
            float r  = 1.f / (1.f + expf(-(gir + ghr)));
            float z  = 1.f / (1.f + expf(-(giz + ghz)));
            float nn = tanhf(gin + r * ghn);
            float hv = (1.f - z) * nn + z * h0f[e];
            h0f[e] = hv; h0b[e] = f2bf(hv);
        }
        grid_sync(cnt, phase);
        // Phase B: gi1(t) = h0 @ Wih1^T and gh0(t+1) = h0 @ Whh0^T (shared A=h0)
        stage_A_lds(h0b, ldsA);
        gemm_strip2(ldsA, Wih1b, gC, wave, lane);
        gemm_strip2(ldsA, Whh0b, gA, wave, lane);
        grid_sync(cnt, phase);
        // Phase C: combine layer1 -> h1(t), record enc_outs (bf16)
        gru_combine(tid, nth, gC, gB, bih1, bhh1, h1f, h1b, enc_outs, t);
        grid_sync(cnt, phase);
    }
}

// ---------------- decoder: persistent, 10 phases per step ----------------
__global__ void __launch_bounds__(TPB) t2s_decoder_kernel(
        const unsigned short* __restrict__ embb,
        const unsigned short* __restrict__ Wi0b, const unsigned short* __restrict__ Wh0b,
        const unsigned short* __restrict__ Wi1b, const unsigned short* __restrict__ Wh1b,
        const float* __restrict__ bi0, const float* __restrict__ bh0,
        const float* __restrict__ bi1, const float* __restrict__ bh1,
        const float* __restrict__ Watt, const float* __restrict__ batt,
        float* __restrict__ h0f, float* __restrict__ h1f,
        unsigned short* __restrict__ h0b, unsigned short* __restrict__ h1b,
        unsigned short* __restrict__ inpb,
        float* __restrict__ gD /*gi0*/, float* __restrict__ gA /*gh0*/,
        float* __restrict__ gB /*gh1*/, float* __restrict__ gC /*gi1*/,
        const unsigned short* __restrict__ enc_outs,
        float* __restrict__ sim, float* __restrict__ colmax, float* __restrict__ colsum,
        float* __restrict__ ctx, int* __restrict__ tok,
        float* __restrict__ out, unsigned* __restrict__ cnt)
{
    __shared__ __align__(16) unsigned short ldsA[BSZ * HID];   // 64 KB
    const int tid  = blockIdx.x * TPB + threadIdx.x;
    const int nth  = NBLK * TPB;
    const int wave = tid >> 5, lane = tid & 31;
    unsigned phase = 0;

    for (int step = 0; step < MAXL; ++step) {
        // P1: gather embedding of previous token
        for (int e = tid; e < BSZ * HID; e += nth) {
            const int b = e >> 9, j = e & 511;
            inpb[e] = embb[tok[b] * HID + j];
        }
        grid_sync(cnt, phase);
        // P2: gi0 / gh0 / gh1 GEMMs (each staged through LDS, every wave 1 strip each)
        stage_A_lds(inpb, ldsA);
        gemm_strip2(ldsA, Wi0b, gD, wave, lane);
        stage_A_lds(h0b, ldsA);
        gemm_strip2(ldsA, Wh0b, gA, wave, lane);
        stage_A_lds(h1b, ldsA);
        gemm_strip2(ldsA, Wh1b, gB, wave, lane);
        grid_sync(cnt, phase);
        // P3: combine layer0
        gru_combine(tid, nth, gD, gA, bi0, bh0, h0f, h0b, (unsigned short*)0, 0);
        grid_sync(cnt, phase);
        // P4: gi1 GEMM
        stage_A_lds(h0b, ldsA);
        gemm_strip2(ldsA, Wi1b, gC, wave, lane);
        grid_sync(cnt, phase);
        // P5: combine layer1 -> out = h1
        gru_combine(tid, nth, gC, gB, bi1, bh1, h1f, h1b, (unsigned short*)0, 0);
        grid_sync(cnt, phase);
        // P6: sim[b,t] = enc_outs[b,t,:] . out[b,:]
        for (int p = tid; p < BSZ * TSEQ; p += nth) {
            const int b = p >> 9, t = p & 511;
            const unsigned short* er = enc_outs + (b * TSEQ + t) * HID;
            const float* o = h1f + b * HID;
            float s = 0.f;
#pragma unroll 8
            for (int h = 0; h < HID; ++h) s += bf2f(er[h]) * o[h];
            sim[p] = s;
        }
        grid_sync(cnt, phase);
        // P7: softmax stats over the BATCH axis (faithful to torch Softmax() on 3D)
        if (tid < TSEQ) {
            float m = -INFINITY;
            for (int b = 0; b < BSZ; ++b) m = fmaxf(m, sim[b * TSEQ + tid]);
            float s = 0.f;
            for (int b = 0; b < BSZ; ++b) s += expf(sim[b * TSEQ + tid] - m);
            colmax[tid] = m; colsum[tid] = s;
        }
        grid_sync(cnt, phase);
        // P8: materialize attn in place
        for (int p = tid; p < BSZ * TSEQ; p += nth) {
            const int t = p & 511;
            sim[p] = expf(sim[p] - colmax[t]) / colsum[t];
        }
        grid_sync(cnt, phase);
        // P9: ctx[b,h] = sum_t attn[b,t] * enc_outs[b,t,h]
        for (int e = tid; e < BSZ * HID; e += nth) {
            const int b = e >> 9, h = e & 511;
            const float* a = sim + b * TSEQ;
            const unsigned short* er = enc_outs + b * TSEQ * HID + h;
            float acc = 0.f;
#pragma unroll 4
            for (int t = 0; t < TSEQ; ++t) acc += a[t] * bf2f(er[t * HID]);
            ctx[e] = acc;
        }
        grid_sync(cnt, phase);
        // P10: logits = [out|ctx] @ Watt^T + batt ; argmax feedback ; emit logits
        if (wave < BSZ) {
            const int b = wave;
            float logit = -INFINITY;
            if (lane < VOC) {
                const float* wr = Watt + lane * (2 * HID);
                const float* o = h1f + b * HID;
                const float* c = ctx + b * HID;
                float s = batt[lane];
#pragma unroll 4
                for (int h = 0; h < HID; ++h) s += o[h] * wr[h] + c[h] * wr[HID + h];
                logit = s;
                out[(b * MAXL + step) * VOC + lane] = s;
            }
            float bv = logit; int bi = lane;
#pragma unroll
            for (int off = 16; off; off >>= 1) {
                float ov = __shfl_down(bv, off, 32);
                int   oi = __shfl_down(bi, off, 32);
                if (ov > bv) { bv = ov; bi = oi; }
            }
            if (lane == 0) tok[b] = bi;
        }
        grid_sync(cnt, phase);
    }
    // hF = final decoder hidden [L=2, B, H], appended after the logits
    for (int e = tid; e < BSZ * HID; e += nth) {
        out[BMV + e]             = h0f[e];
        out[BMV + BSZ * HID + e] = h1f[e];
    }
}

// ---------------- host launch ----------------
extern "C" void kernel_launch(void* const* d_in, const int* in_sizes, int n_in,
                              void* d_out, int out_size, void* d_ws, size_t ws_size,
                              hipStream_t stream) {
    const float* x        = (const float*)d_in[0];
    const float* emb      = (const float*)d_in[1];
    const float* eWih0    = (const float*)d_in[2];
    const float* eWhh0    = (const float*)d_in[3];
    const float* ebih0    = (const float*)d_in[4];
    const float* ebhh0    = (const float*)d_in[5];
    const float* eWih1    = (const float*)d_in[6];
    const float* eWhh1    = (const float*)d_in[7];
    const float* ebih1    = (const float*)d_in[8];
    const float* ebhh1    = (const float*)d_in[9];
    const float* dWih0    = (const float*)d_in[10];
    const float* dWhh0    = (const float*)d_in[11];
    const float* dbih0    = (const float*)d_in[12];
    const float* dbhh0    = (const float*)d_in[13];
    const float* dWih1    = (const float*)d_in[14];
    const float* dWhh1    = (const float*)d_in[15];
    const float* dbih1    = (const float*)d_in[16];
    const float* dbhh1    = (const float*)d_in[17];
    const float* Watt     = (const float*)d_in[18];
    const float* batt     = (const float*)d_in[19];

    char* base = (char*)d_ws;
    size_t off = 0;
    auto take = [&](size_t bytes) -> char* {
        char* p = base + off;
        off += (bytes + 255) & ~(size_t)255;
        return p;
    };
    unsigned*       sync_cnt = (unsigned*)take(256);
    unsigned short* Wb       = (unsigned short*)take((size_t)7 * WSTR * 2);
    unsigned short* embb     = (unsigned short*)take((size_t)VOC * HID * 2);
    unsigned short* enc_outs = (unsigned short*)take((size_t)BSZ * TSEQ * HID * 2);
    float*          h0f      = (float*)take((size_t)BSZ * HID * 4);
    float*          h1f      = (float*)take((size_t)BSZ * HID * 4);
    unsigned short* h0b      = (unsigned short*)take((size_t)BSZ * HID * 2);
    unsigned short* h1b      = (unsigned short*)take((size_t)BSZ * HID * 2);
    unsigned short* inpb     = (unsigned short*)take((size_t)BSZ * HID * 2);
    float*          gA       = (float*)take((size_t)BSZ * G3H * 4);
    float*          gB       = (float*)take((size_t)BSZ * G3H * 4);
    float*          gC       = (float*)take((size_t)BSZ * G3H * 4);
    float*          gD       = (float*)take((size_t)BSZ * G3H * 4);
    float*          sim      = (float*)take((size_t)BSZ * TSEQ * 4);
    float*          colmax   = (float*)take((size_t)TSEQ * 4);
    float*          colsum   = (float*)take((size_t)TSEQ * 4);
    float*          ctx      = (float*)take((size_t)BSZ * HID * 4);
    int*            tok      = (int*)take(256);

    hipMemsetAsync(sync_cnt, 0, 4, stream);
    // bf16 weight order: 0 enc_Whh0, 1 enc_Wih1, 2 enc_Whh1,
    //                    3 dec_Wih0, 4 dec_Whh0, 5 dec_Wih1, 6 dec_Whh1
    t2s_init_kernel<<<256, TPB, 0, stream>>>(
        eWhh0, eWih1, eWhh1, dWih0, dWhh0, dWih1, dWhh1, emb,
        Wb, embb, h0f, h1f, h0b, h1b, gA, tok);

    t2s_encoder_kernel<<<NBLK, TPB, 0, stream>>>(
        x, eWih0, ebih0, ebhh0,
        Wb + 0 * WSTR, Wb + 1 * WSTR, Wb + 2 * WSTR,
        ebih1, ebhh1,
        h0f, h1f, h0b, h1b, gA, gB, gC, enc_outs, sync_cnt);

    hipMemsetAsync(sync_cnt, 0, 4, stream);
    t2s_decoder_kernel<<<NBLK, TPB, 0, stream>>>(
        embb,
        Wb + 3 * WSTR, Wb + 4 * WSTR, Wb + 5 * WSTR, Wb + 6 * WSTR,
        dbih0, dbhh0, dbih1, dbhh1, Watt, batt,
        h0f, h1f, h0b, h1b, inpb,
        gD, gA, gB, gC, enc_outs,
        sim, colmax, colsum, ctx, tok,
        (float*)d_out, sync_cnt);
}
